// VolumetricRenderer_16312285790431
// MI455X (gfx1250) — matrix operand order repaired and checked by reading, not compiled, not run on hardware
//
#include <hip/hip_runtime.h>
#include <math.h>

// ---------------------------------------------------------------------------
// Fused NeRF-style volumetric renderer for gfx1250 (MI455X).
//   - one wave32 per ray (64 samples = 4 tiles of 16 rows)
//   - layer 1 (32->128) via v_wmma_f32_16x16x32_f16 (8 WMMAs / tile,
//     t-loop kept at unroll-2 so only 2 B-fragments + 2 accumulators are
//     live -> no scratch spills under the 256-VGPR cap)
//   - layer 2 (128->4) fused as VALU MAC + wave butterfly reduction
//   - compositing done in-register, no workspace traffic
// ---------------------------------------------------------------------------

typedef __attribute__((ext_vector_type(16))) _Float16 v16h;
typedef __attribute__((ext_vector_type(8)))  float    v8f;

#define DFEAT        32
#define PLANE_RES    128
#define H_R          128
#define W_R          128
#define HW           (H_R * W_R)
#define CH_SLAB      (PLANE_RES * PLANE_RES)   // 16384 floats per channel slice
#define PLANE_STRIDE (DFEAT * CH_SLAB)         // 524288 floats per plane

__device__ __forceinline__ float sigmoid_f(float x) {
    return 1.0f / (1.0f + __expf(-x));
}

__device__ __forceinline__ void grid_prep(float gx, float gy,
                                          int& o00, int& o01, int& o10, int& o11,
                                          float& w00, float& w01, float& w10, float& w11) {
    // align_corners=True bilinear over a 128x128 plane, inputs in [-1,1]
    float x = fminf(fmaxf((gx + 1.0f) * 0.5f * 127.0f, 0.0f), 127.0f);
    float y = fminf(fmaxf((gy + 1.0f) * 0.5f * 127.0f, 0.0f), 127.0f);
    int x0 = (int)floorf(x);
    int y0 = (int)floorf(y);
    int x1 = min(x0 + 1, 127);
    int y1 = min(y0 + 1, 127);
    float wx = x - (float)x0;
    float wy = y - (float)y0;
    o00 = y0 * 128 + x0;  o01 = y0 * 128 + x1;
    o10 = y1 * 128 + x0;  o11 = y1 * 128 + x1;
    w00 = (1.0f - wx) * (1.0f - wy);
    w01 = wx * (1.0f - wy);
    w10 = (1.0f - wx) * wy;
    w11 = wx * wy;
}

__global__ void __launch_bounds__(256, 4)
nerf_render_wmma(const float* __restrict__ planes,   // (1,3,32,128,128)
                 const float* __restrict__ intr,     // (1,4,3,3)
                 const float* __restrict__ extr,     // (1,4,4,4)
                 const float* __restrict__ W1,       // (32,128)
                 const float* __restrict__ b1,       // (128,)
                 const float* __restrict__ W2,       // (128,4)
                 const float* __restrict__ b2,       // (4,)
                 const int*   __restrict__ cidx,     // (C,)
                 float*       __restrict__ out,      // (1,C,3,128,128)
                 int n_rays)
{
    // ---- stage weights in LDS (W1 transposed to [col][k], fp16) -----------
    __shared__ __align__(64) _Float16 W1T[128 * 32];  // 8 KB
    __shared__ float  b1s[128];
    __shared__ float4 W2s[128];
    __shared__ float4 b2s;

    const int tid = threadIdx.x;
    for (int idx = tid; idx < 128 * 32; idx += 256) {
        int col = idx >> 5;          // hidden column 0..127
        int k   = idx & 31;          // input feature 0..31
        W1T[idx] = (_Float16)W1[k * 128 + col];
    }
    if (tid < 128) {
        b1s[tid] = b1[tid];
        W2s[tid] = make_float4(W2[tid * 4 + 0], W2[tid * 4 + 1],
                               W2[tid * 4 + 2], W2[tid * 4 + 3]);
    }
    if (tid == 0) b2s = make_float4(b2[0], b2[1], b2[2], b2[3]);
    __syncthreads();

    const int wave = blockIdx.x * 8 + (tid >> 5);   // uniform per wave
    if (wave >= n_rays) return;

    const int lane = tid & 31;
    const int r    = lane & 15;   // row/column index within 16
    const int h    = lane >> 4;   // half of the wave

    // ---- ray setup (uniform within the wave) ------------------------------
    const int cam = wave / HW;
    const int pix = wave - cam * HW;
    const int vpix = pix >> 7;          // row
    const int upix = pix & (W_R - 1);   // col
    const int ci   = cidx[cam];

    const float fx = intr[ci * 9 + 0] * 0.5f;
    const float cx = intr[ci * 9 + 2] * 0.5f;
    const float fy = intr[ci * 9 + 4] * 0.5f;
    const float cy = intr[ci * 9 + 5] * 0.5f;

    const float* E = extr + ci * 16;    // row-major 4x4
    const float dcx = ((float)upix - cx) / fx;
    const float dcy = ((float)vpix - cy) / fy;
    // dir = R^T * dir_cam ; origin = -R^T * t
    float dx = E[0] * dcx + E[4] * dcy + E[8];
    float dy = E[1] * dcx + E[5] * dcy + E[9];
    float dz = E[2] * dcx + E[6] * dcy + E[10];
    const float inv = rsqrtf(dx * dx + dy * dy + dz * dz);
    dx *= inv; dy *= inv; dz *= inv;
    const float ox = -(E[0] * E[3] + E[4] * E[7] + E[8]  * E[11]);
    const float oy = -(E[1] * E[3] + E[5] * E[7] + E[9]  * E[11]);
    const float oz = -(E[2] * E[3] + E[6] * E[7] + E[10] * E[11]);

    const float4 b2v = b2s;
    const v8f czero = {};

    float T = 1.0f, crr = 0.0f, cgg = 0.0f, cbb = 0.0f;

    for (int ts = 0; ts < 4; ++ts) {
        // ---- this lane's sample: row r of the tile ------------------------
        const int   s    = ts * 16 + r;
        const float tval = 0.5f + 3.0f * (float)s * (1.0f / 63.0f);
        const float px   = ox + tval * dx;
        const float py   = oy + tval * dy;
        const float pz   = oz + tval * dz;
        const float pnx  = fminf(fmaxf(px * (1.0f / 3.5f), -1.0f), 1.0f);
        const float pny  = fminf(fmaxf(py * (1.0f / 3.5f), -1.0f), 1.0f);
        const float pnz  = fminf(fmaxf(pz * (1.0f / 3.5f), -1.0f), 1.0f);

        int oA00, oA01, oA10, oA11, oB00, oB01, oB10, oB11, oC00, oC01, oC10, oC11;
        float wA00, wA01, wA10, wA11, wB00, wB01, wB10, wB11, wC00, wC01, wC10, wC11;
        grid_prep(pnx, pny, oA00, oA01, oA10, oA11, wA00, wA01, wA10, wA11); // xy
        grid_prep(pnx, pnz, oB00, oB01, oB10, oB11, wB00, wB01, wB10, wB11); // xz
        grid_prep(pny, pnz, oC00, oC01, oC10, oC11, wC00, wC01, wC10, wC11); // yz

        // ---- build A fragment: 16x32 fp16, k = 16h + j order -------------------
        // half 0 lanes supply K in {0..15}; half 1 K in {16..31}
        v16h a;
#pragma unroll
        for (int j = 0; j < 16; ++j) {
            const int ch = j + 16 * h;
            const float* P0 = planes + ch * CH_SLAB;
            const float* P1 = P0 + PLANE_STRIDE;
            const float* P2 = P1 + PLANE_STRIDE;
            float f = wA00 * P0[oA00] + wA01 * P0[oA01] + wA10 * P0[oA10] + wA11 * P0[oA11]
                    + wB00 * P1[oB00] + wB01 * P1[oB01] + wB10 * P1[oB10] + wB11 * P1[oB11]
                    + wC00 * P2[oC00] + wC01 * P2[oC01] + wC10 * P2[oC10] + wC11 * P2[oC11];
            a[j] = (_Float16)f;
        }

        // ---- layer 1 via WMMA, layer 2 fused per column tile --------------
        float4 part[8];
#pragma unroll
        for (int i = 0; i < 8; ++i) part[i] = make_float4(0.f, 0.f, 0.f, 0.f);

        // unroll-2: only 2 B fragments + 2 accumulators live at once; the
        // B fragment is re-read from LDS every iteration (2x ds_load_b128)
        // instead of being spilled to scratch under register pressure.
#pragma unroll 2
        for (int t = 0; t < 8; ++t) {
            // B fragment: column (t*16 + r), K = j + 16*h, contiguous in W1T
            const v16h bm = *(const v16h*)(&W1T[(t * 16 + r) * 32 + 16 * h]);
            v8f acc = __builtin_amdgcn_wmma_f32_16x16x32_f16(
                false, a, false, bm, (short)0, czero, false, false);
            const float  bb = b1s[t * 16 + r];
            const float4 w2 = W2s[t * 16 + r];
#pragma unroll
            for (int i = 0; i < 8; ++i) {
                // acc[i] = H[row = i + 8h][hidden k = t*16 + r]
                float hk = fmaxf(acc[i] + bb, 0.0f);
                part[i].x += hk * w2.x;
                part[i].y += hk * w2.y;
                part[i].z += hk * w2.z;
                part[i].w += hk * w2.w;
            }
        }

        // ---- reduce hidden dim across the 16 lanes of each half -----------
#pragma unroll
        for (int m = 1; m < 16; m <<= 1) {
#pragma unroll
            for (int i = 0; i < 8; ++i) {
                part[i].x += __shfl_xor(part[i].x, m, 32);
                part[i].y += __shfl_xor(part[i].y, m, 32);
                part[i].z += __shfl_xor(part[i].z, m, 32);
                part[i].w += __shfl_xor(part[i].w, m, 32);
            }
        }

        // ---- sequential alpha compositing over this tile's 16 samples -----
        // part[i] on half h holds sample row (i + 8h); the other half's row
        // is fetched on the fly with an xor-16 shuffle (no lo/hi arrays).
#pragma unroll
        for (int p = 0; p < 16; ++p) {
            const int    i    = p & 7;
            const float4 mine = part[i];
            float4 oth;
            oth.x = __shfl_xor(mine.x, 16, 32);
            oth.y = __shfl_xor(mine.y, 16, 32);
            oth.z = __shfl_xor(mine.z, 16, 32);
            oth.w = __shfl_xor(mine.w, 16, 32);
            const bool useMine = ((p >> 3) == h);   // per-lane select, same value
            const float vx = useMine ? mine.x : oth.x;
            const float vy = useMine ? mine.y : oth.y;
            const float vz = useMine ? mine.z : oth.z;
            const float vw = useMine ? mine.w : oth.w;

            const float rr = sigmoid_f(vx + b2v.x);
            const float rg = sigmoid_f(vy + b2v.y);
            const float rb = sigmoid_f(vz + b2v.z);
            const float sx = vw + b2v.w;
            const float sg = (sx > 20.0f) ? sx : __logf(1.0f + __expf(sx)); // softplus
            const int   sc = ts * 16 + p;
            const float delta = (sc < 63) ? (3.0f / 63.0f) : 1.0e10f;
            const float alpha = 1.0f - __expf(-sg * delta);
            const float w = alpha * T;
            crr += w * rr;
            cgg += w * rg;
            cbb += w * rb;
            T *= (1.0f - alpha + 1e-10f);
        }
    }

    if (lane == 0) {
        out[(cam * 3 + 0) * HW + pix] = crr;
        out[(cam * 3 + 1) * HW + pix] = cgg;
        out[(cam * 3 + 2) * HW + pix] = cbb;
    }
}

extern "C" void kernel_launch(void* const* d_in, const int* in_sizes, int n_in,
                              void* d_out, int out_size, void* d_ws, size_t ws_size,
                              hipStream_t stream) {
    const float* planes = (const float*)d_in[0];
    const float* intr   = (const float*)d_in[1];
    const float* extr   = (const float*)d_in[2];
    const float* W1     = (const float*)d_in[3];
    const float* b1     = (const float*)d_in[4];
    const float* W2     = (const float*)d_in[5];
    const float* b2     = (const float*)d_in[6];
    const int*   cidx   = (const int*)d_in[7];
    float*       out    = (float*)d_out;

    const int C      = in_sizes[7];       // number of selected cameras (2)
    const int n_rays = C * HW;            // 32768 rays, one wave32 each
    const int waves_per_block = 8;        // 256 threads
    const int blocks = (n_rays + waves_per_block - 1) / waves_per_block;

    nerf_render_wmma<<<blocks, 256, 0, stream>>>(
        planes, intr, extr, W1, b1, W2, b2, cidx, out, n_rays);
}
